// GraphAttentionLayer_12214886990142
// MI455X (gfx1250) — matrix operand order repaired and checked by reading, not compiled, not run on hardware
//
#include <hip/hip_runtime.h>

#define BATCH 4
#define NNODE 2048
#define INF_  256
#define HEADS 4
#define FDIM  64
#define HF    256   // HEADS*FDIM
#define NEG_SLOPE 0.2f

typedef __attribute__((ext_vector_type(8)))  float  v8f;
typedef __attribute__((ext_vector_type(16))) __bf16 v16bf;
typedef __attribute__((ext_vector_type(8)))  __bf16 v8bf;

union BF16x16 { v16bf v; uint4 q[2]; };
union BF16x8  { v8bf  v; uint4 q; };

// Native bf16 convert: backend emits v_cvt_*bf16* on gfx1250 (RNE).
__device__ __forceinline__ __bf16 tobf(float f) { return (__bf16)f; }

// ---------------------------------------------------------------------------
// K0: WT[n][k] = bf16(W[k][n])   (256x256)
// ---------------------------------------------------------------------------
__global__ __launch_bounds__(256) void k_wconv(const float* __restrict__ W,
                                               __bf16* __restrict__ WT) {
  int n = blockIdx.x, k = threadIdx.x;
  WT[n * INF_ + k] = tobf(W[k * HF + n]);
}

// ---------------------------------------------------------------------------
// K1: h = x @ W  (8192x256x256), bf16 WMMA, f32 accum.
// Writes h (f32, row-major) and hbT (bf16, [b][head][f][j]).
// Block = 8 waves; wave w owns cols [w*32, w*32+32); block per 16-row tile.
// ---------------------------------------------------------------------------
__global__ __launch_bounds__(256) void k_gemm_xw(const float* __restrict__ x,
                                                 const __bf16* __restrict__ WT,
                                                 float* __restrict__ h,
                                                 __bf16* __restrict__ hbT) {
  const int wave = threadIdx.x >> 5;
  const int lane = threadIdx.x & 31;
  const int half = lane >> 4;
  const int lm   = lane & 15;
  const int i0   = blockIdx.x * 16;          // global row tile (of 8192)
  const int n0   = wave * 32;                // column tile base

  v8f c0 = {}; v8f c1 = {};
  const float* xrow = x + (size_t)(i0 + lm) * INF_;
  const __bf16* wt0base = WT + (size_t)(n0 + lm) * INF_;
  const __bf16* wt1base = WT + (size_t)(n0 + 16 + lm) * INF_;

  for (int kt = 0; kt < INF_; kt += 32) {
    // ---- A tile 16x32 bf16: lane row = lm; k = kt+half*8+{0..7} U kt+16+half*8+{0..7}
    BF16x16 A;
    const int ka = kt + half * 8;
    float4 f0 = *(const float4*)(xrow + ka);
    float4 f1 = *(const float4*)(xrow + ka + 4);
    float4 f2 = *(const float4*)(xrow + ka + 16);
    float4 f3 = *(const float4*)(xrow + ka + 20);
    A.v[0]=tobf(f0.x);  A.v[1]=tobf(f0.y);  A.v[2]=tobf(f0.z);  A.v[3]=tobf(f0.w);
    A.v[4]=tobf(f1.x);  A.v[5]=tobf(f1.y);  A.v[6]=tobf(f1.z);  A.v[7]=tobf(f1.w);
    A.v[8]=tobf(f2.x);  A.v[9]=tobf(f2.y);  A.v[10]=tobf(f2.z); A.v[11]=tobf(f2.w);
    A.v[12]=tobf(f3.x); A.v[13]=tobf(f3.y); A.v[14]=tobf(f3.z); A.v[15]=tobf(f3.w);

    // ---- B tiles 32x16 bf16: lane col = n; k = kt+half*8+{0..7} U kt+16+half*8+{0..7}
    const int kb = kt + half * 8;
    BF16x16 B0, B1;
    const uint4* p0 = (const uint4*)(wt0base + kb);
    const uint4* p1 = (const uint4*)(wt1base + kb);
    B0.q[0] = p0[0]; B0.q[1] = p0[2];
    B1.q[0] = p1[0]; B1.q[1] = p1[2];

    c0 = __builtin_amdgcn_wmma_f32_16x16x32_bf16(false, A.v, false, B0.v, (short)0, c0, false, false);
    c1 = __builtin_amdgcn_wmma_f32_16x16x32_bf16(false, A.v, false, B1.v, (short)0, c1, false, false);
  }

  // ---- store: C lane col = n, rows m = r + 8*half
  const int b  = i0 >> 11;                   // i0 / NNODE
  const int j0 = (i0 & (NNODE - 1)) + 8 * half;
  {
    const int n = n0 + lm, head = n >> 6, f = n & 63;
    BF16x8 t;
    #pragma unroll
    for (int r = 0; r < 8; ++r) {
      h[(size_t)(i0 + 8 * half + r) * HF + n] = c0[r];
      t.v[r] = tobf(c0[r]);
    }
    *(uint4*)(hbT + ((size_t)((b * HEADS + head) * FDIM) + f) * NNODE + j0) = t.q;
  }
  {
    const int n = n0 + 16 + lm, head = n >> 6, f = n & 63;
    BF16x8 t;
    #pragma unroll
    for (int r = 0; r < 8; ++r) {
      h[(size_t)(i0 + 8 * half + r) * HF + n] = c1[r];
      t.v[r] = tobf(c1[r]);
    }
    *(uint4*)(hbT + ((size_t)((b * HEADS + head) * FDIM) + f) * NNODE + j0) = t.q;
  }
}

// ---------------------------------------------------------------------------
// K1b: s1[rh] = h[row, hd*64+:] . a[:64]; s2 with a[64:]. rh = row*H + hd.
// ---------------------------------------------------------------------------
__global__ __launch_bounds__(256) void k_scores(const float* __restrict__ h,
                                                const float* __restrict__ a,
                                                float* __restrict__ s1,
                                                float* __restrict__ s2) {
  int tid = blockIdx.x * blockDim.x + threadIdx.x;      // 0 .. B*N*H-1
  int hd = tid & (HEADS - 1);
  int row = tid >> 2;
  const float* hp = h + (size_t)row * HF + hd * FDIM;
  float acc1 = 0.f, acc2 = 0.f;
  #pragma unroll 8
  for (int f = 0; f < FDIM; ++f) {
    float v = hp[f];
    acc1 += v * a[f];
    acc2 += v * a[FDIM + f];
  }
  s1[tid] = acc1;
  s2[tid] = acc2;
}

// ---------------------------------------------------------------------------
// K2: D[b,j,h] += sum_{i in slice} adj[b,i,j] * exp(lrelu(s1[b,i,h]+s2[b,j,h]))
// grid = (B*8 j-blocks, 16 i-slices); thread <-> j (coalesced adj rows).
// ---------------------------------------------------------------------------
__global__ __launch_bounds__(256) void k_colsum(const int* __restrict__ adj,
                                                const float* __restrict__ s1,
                                                const float* __restrict__ s2,
                                                float* __restrict__ D) {
  const int b = blockIdx.x >> 3;
  const int j = ((blockIdx.x & 7) << 8) + threadIdx.x;
  const int i0 = blockIdx.y * 128;

  float4 t2 = *(const float4*)(s2 + ((size_t)(b * NNODE + j)) * HEADS);
  float d0 = 0.f, d1 = 0.f, d2 = 0.f, d3 = 0.f;

  const int*   adjp = adj + ((size_t)b * NNODE + i0) * NNODE + j;
  const float* s1p  = s1 + ((size_t)(b * NNODE + i0)) * HEADS;

  for (int i = 0; i < 128; ++i) {
    int av = adjp[(size_t)i * NNODE];
    if (av != 0) {
      float4 s1v = *(const float4*)(s1p + i * 4);   // uniform -> scalar load
      float t, e;
      t = s1v.x + t2.x; e = fmaxf(t, NEG_SLOPE * t); d0 += __expf(e);
      t = s1v.y + t2.y; e = fmaxf(t, NEG_SLOPE * t); d1 += __expf(e);
      t = s1v.z + t2.z; e = fmaxf(t, NEG_SLOPE * t); d2 += __expf(e);
      t = s1v.w + t2.w; e = fmaxf(t, NEG_SLOPE * t); d3 += __expf(e);
    }
  }
  float* dp = D + ((size_t)(b * NNODE + j)) * HEADS;
  atomicAdd(dp + 0, d0);
  atomicAdd(dp + 1, d1);
  atomicAdd(dp + 2, d2);
  atomicAdd(dp + 3, d3);
}

// ---------------------------------------------------------------------------
// K2p: pack[(b*H+h)*N + j] = { s2[b,j,h], 1/D[b,j,h] }
// ---------------------------------------------------------------------------
__global__ __launch_bounds__(256) void k_pack(const float* __restrict__ s2,
                                              const float* __restrict__ D,
                                              float2* __restrict__ pack) {
  int tid = blockIdx.x * blockDim.x + threadIdx.x;      // 0 .. B*H*N-1
  int b = tid >> 13;
  int rem = tid & 8191;
  int hd = rem >> 11;
  int j = rem & (NNODE - 1);
  size_t src = (size_t)(b * NNODE + j) * HEADS + hd;
  pack[((size_t)(b * HEADS + hd)) * NNODE + j] = make_float2(s2[src], 1.0f / D[src]);
}

// ---------------------------------------------------------------------------
// K3: out[b,i,head*64+f] = sum_j A[i,j] * G[j,f],
//   A[i,j] = adj[b,i,j] * exp(lrelu(s1[i]+s2[j])) * (1/D[j]),  G = hbT (bf16)
// One wave per (b,head,i-tile); 4 WMMA accumulators cover all 64 features.
// ---------------------------------------------------------------------------
__global__ __launch_bounds__(256) void k_aggr(const int* __restrict__ adj,
                                              const float* __restrict__ s1,
                                              const float2* __restrict__ pack,
                                              const __bf16* __restrict__ hbT,
                                              float* __restrict__ out) {
  const int w     = blockIdx.x * 8 + (threadIdx.x >> 5);
  const int lane  = threadIdx.x & 31;
  const int itile = w & 127;
  const int bh    = w >> 7;             // 0..15
  const int b     = bh >> 2;
  const int head  = bh & 3;
  const int half  = lane >> 4;
  const int lm    = lane & 15;
  const int i0    = itile * 16;
  const int irow  = i0 + lm;            // A-operand row owned by this lane

  const float s1m = s1[((size_t)(b * NNODE) + irow) * HEADS + head];
  const int* adjrow = adj + ((size_t)b * NNODE + irow) * NNODE;
  const float2* pk  = pack + (size_t)bh * NNODE;
  const __bf16* gbase = hbT + (size_t)bh * FDIM * NNODE;

  v8f c0 = {}, c1 = {}, c2 = {}, c3 = {};

  for (int kt = 0; kt < NNODE; kt += 32) {
    const int ja = kt + half * 8;
    const int jb = ja + 16;

    __builtin_prefetch(adjrow + ja + 64, 0, 1);   // global_prefetch next chunk

    int4 A0 = *(const int4*)(adjrow + ja);
    int4 A1 = *(const int4*)(adjrow + ja + 4);
    int4 A2 = *(const int4*)(adjrow + jb);
    int4 A3 = *(const int4*)(adjrow + jb + 4);

    const float4* pa = (const float4*)(pk + ja);   // {s2,scale} pairs
    const float4* pb = (const float4*)(pk + jb);
    float4 pa0 = pa[0], pa1 = pa[1], pa2 = pa[2], pa3 = pa[3];
    float4 pb0 = pb[0], pb1 = pb[1], pb2 = pb[2], pb3 = pb[3];

    BF16x16 Am;
    #define ENT(av, sj, sc) ({ float t_ = s1m + (sj);                       \
                               float e_ = fmaxf(t_, NEG_SLOPE * t_);        \
                               tobf(__expf(e_) * (sc) * (float)(av)); })
    Am.v[0]  = ENT(A0.x, pa0.x, pa0.y);
    Am.v[1]  = ENT(A0.y, pa0.z, pa0.w);
    Am.v[2]  = ENT(A0.z, pa1.x, pa1.y);
    Am.v[3]  = ENT(A0.w, pa1.z, pa1.w);
    Am.v[4]  = ENT(A1.x, pa2.x, pa2.y);
    Am.v[5]  = ENT(A1.y, pa2.z, pa2.w);
    Am.v[6]  = ENT(A1.z, pa3.x, pa3.y);
    Am.v[7]  = ENT(A1.w, pa3.z, pa3.w);
    Am.v[8]  = ENT(A2.x, pb0.x, pb0.y);
    Am.v[9]  = ENT(A2.y, pb0.z, pb0.w);
    Am.v[10] = ENT(A2.z, pb1.x, pb1.y);
    Am.v[11] = ENT(A2.w, pb1.z, pb1.w);
    Am.v[12] = ENT(A3.x, pb2.x, pb2.y);
    Am.v[13] = ENT(A3.y, pb2.z, pb2.w);
    Am.v[14] = ENT(A3.z, pb3.x, pb3.y);
    Am.v[15] = ENT(A3.w, pb3.z, pb3.w);
    #undef ENT

    // B tiles: lane col = ft*16+lm; k = kt+half*8+{0..7} U kt+16+half*8+{0..7}
    const int kb2 = kt + half * 8;
    BF16x16 Bm;
    const uint4* g0 = (const uint4*)(gbase + (size_t)(0 * 16 + lm) * NNODE + kb2);
    Bm.q[0] = g0[0]; Bm.q[1] = g0[2];
    c0 = __builtin_amdgcn_wmma_f32_16x16x32_bf16(false, Am.v, false, Bm.v, (short)0, c0, false, false);
    const uint4* g1 = (const uint4*)(gbase + (size_t)(1 * 16 + lm) * NNODE + kb2);
    Bm.q[0] = g1[0]; Bm.q[1] = g1[2];
    c1 = __builtin_amdgcn_wmma_f32_16x16x32_bf16(false, Am.v, false, Bm.v, (short)0, c1, false, false);
    const uint4* g2 = (const uint4*)(gbase + (size_t)(2 * 16 + lm) * NNODE + kb2);
    Bm.q[0] = g2[0]; Bm.q[1] = g2[2];
    c2 = __builtin_amdgcn_wmma_f32_16x16x32_bf16(false, Am.v, false, Bm.v, (short)0, c2, false, false);
    const uint4* g3 = (const uint4*)(gbase + (size_t)(3 * 16 + lm) * NNODE + kb2);
    Bm.q[0] = g3[0]; Bm.q[1] = g3[2];
    c3 = __builtin_amdgcn_wmma_f32_16x16x32_bf16(false, Am.v, false, Bm.v, (short)0, c3, false, false);
  }

  // store: out[b][i0+8*half+r][head*64 + ft*16 + lm]
  float* obase = out + ((size_t)(b * NNODE) + i0 + 8 * half) * HF + head * FDIM + lm;
  #pragma unroll
  for (int r = 0; r < 8; ++r) {
    obase[(size_t)r * HF + 0]  = c0[r];
    obase[(size_t)r * HF + 16] = c1[r];
    obase[(size_t)r * HF + 32] = c2[r];
    obase[(size_t)r * HF + 48] = c3[r];
  }
}

// ---------------------------------------------------------------------------
extern "C" void kernel_launch(void* const* d_in, const int* in_sizes, int n_in,
                              void* d_out, int out_size, void* d_ws, size_t ws_size,
                              hipStream_t stream) {
  const float* x   = (const float*)d_in[0];   // (4,2048,256) f32
  const int*   adj = (const int*)d_in[1];     // (4,2048,2048) i32
  const float* W   = (const float*)d_in[2];   // (256,256) f32
  const float* a   = (const float*)d_in[3];   // (128,1) f32
  float* out = (float*)d_out;                 // (4,2048,256) f32

  char* ws = (char*)d_ws;
  const size_t ROWS = (size_t)BATCH * NNODE;          // 8192
  float*  h    = (float*)(ws);                                          // 8 MB
  __bf16* hbT  = (__bf16*)(ws + ROWS * HF * 4);                         // 4 MB
  char* p = ws + ROWS * HF * 4 + ROWS * HF * 2;
  float* s1   = (float*)p;            p += ROWS * HEADS * 4;            // 128 KB
  float* s2   = (float*)p;            p += ROWS * HEADS * 4;            // 128 KB
  float* D    = (float*)p;            p += ROWS * HEADS * 4;            // 128 KB
  float2* pk  = (float2*)p;           p += ROWS * HEADS * 8;            // 256 KB
  __bf16* WT  = (__bf16*)p;                                             // 128 KB

  k_wconv  <<<HF, INF_, 0, stream>>>(W, WT);
  k_gemm_xw<<<(unsigned)(ROWS / 16), 256, 0, stream>>>(x, WT, h, hbT);
  k_scores <<<(unsigned)(ROWS * HEADS / 256), 256, 0, stream>>>(h, a, s1, s2);
  hipMemsetAsync(D, 0, ROWS * HEADS * 4, stream);
  k_colsum <<<dim3(BATCH * 8, 16), 256, 0, stream>>>(adj, s1, s2, D);
  k_pack   <<<(unsigned)(ROWS * HEADS / 256), 256, 0, stream>>>(s2, D, pk);
  k_aggr   <<<(unsigned)(BATCH * HEADS * (NNODE / 16) / 8), 256, 0, stream>>>(adj, s1, pk, hbT, out);
}
